// VectorQuantizer_38268158607611
// MI455X (gfx1250) — matrix operand order repaired and checked by reading, not compiled, not run on hardware
//
#include <hip/hip_runtime.h>

#define D 512
#define KCB 8192
#define NTOK 16384
#define DECAYF 0.99f
#define OMDF 0.01f
#define EPSF 1e-5f

typedef __attribute__((ext_vector_type(16))) __bf16 v16bf;
typedef __attribute__((ext_vector_type(8)))  float  v8f;
typedef __attribute__((ext_vector_type(4)))  unsigned int u32x4;
typedef int vq_i32x4 __attribute__((vector_size(16)));   // matches builtin's V4i

union FragBF { v16bf v; u32x4 q[2]; };
union Pack16 { unsigned short u[16]; u32x4 q[2]; };

#if defined(__has_builtin)
#if __has_builtin(__builtin_amdgcn_global_load_async_to_lds_b128)
#define VQ_HAVE_ASYNC_LDS 1
#endif
#endif

#define VQ_AS_GLOBAL __attribute__((address_space(1)))
#define VQ_AS_LOCAL  __attribute__((address_space(3)))

__device__ __forceinline__ void vq_wait_async() {
#if defined(__has_builtin) && __has_builtin(__builtin_amdgcn_s_wait_asynccnt)
  __builtin_amdgcn_s_wait_asynccnt(0);
#else
  asm volatile("s_wait_asynccnt 0x0" ::: "memory");
#endif
}

// ---------------------------------------------------------------------------
// Row L2-normalize + convert to bf16.  One wave (32 lanes) per row of 512.
// ---------------------------------------------------------------------------
__global__ __launch_bounds__(256) void vq_normalize_bf16(
    const float* __restrict__ src, unsigned short* __restrict__ dst, int nrows) {
  const int lane = threadIdx.x & 31;
  const int row  = blockIdx.x * 8 + (threadIdx.x >> 5);
  if (row >= nrows) return;
  const float* s = src + (size_t)row * D + lane * 16;
  float v[16];
  float ss = 0.f;
#pragma unroll
  for (int i = 0; i < 16; ++i) { v[i] = s[i]; ss += v[i] * v[i]; }
#pragma unroll
  for (int off = 16; off; off >>= 1) ss += __shfl_xor(ss, off, 32);
  const float scale = rsqrtf(ss + 1e-12f);
  Pack16 p;
#pragma unroll
  for (int i = 0; i < 16; ++i) {
    unsigned int u = __float_as_uint(v[i] * scale);   // round-to-nearest-even bf16
    p.u[i] = (unsigned short)((u + 0x7FFFu + ((u >> 16) & 1u)) >> 16);
  }
  u32x4* dq = (u32x4*)(dst + (size_t)row * D + lane * 16);
  dq[0] = p.q[0];
  dq[1] = p.q[1];
}

// ---------------------------------------------------------------------------
// Fused cosine-sim GEMM + row argmax.
// Block = 256 threads = 8 waves; each wave owns 16 tokens (A in 128 VGPRs,
// resident for the whole K sweep).  The block stages 16-code x 512-D bf16
// tiles (16 KB, contiguous) in a double-buffered LDS region via CDNA5
// async-to-LDS DMA, overlapping the next tile's copy with current WMMAs.
// ---------------------------------------------------------------------------
__global__ __launch_bounds__(256) void vq_argmax_wmma(
    const unsigned short* __restrict__ xn,   // [NTOK][D] bf16
    const unsigned short* __restrict__ cbn,  // [KCB][D]  bf16
    int* __restrict__ idx_i, float* __restrict__ idx_f) {
#if VQ_HAVE_ASYNC_LDS
  __shared__ unsigned short tile[2][16 * D];       // 2 x 16 KB double buffer
#else
  __shared__ unsigned short tile[1][16 * D];       // 16 KB
#endif
  const int tid  = threadIdx.x;
  const int lane = tid & 31;
  const int m    = lane & 15;
  const int hi   = lane >> 4;
  const int rowBase = (blockIdx.x * 8 + (tid >> 5)) * 16;

  // A fragments: 16 k-steps covering D=512.  Lane l<16 holds row m, K-chunks
  // {d0..d0+7, d0+16..d0+23}; lanes 16..31 the +8 shifted chunks (ISA 7.12.2).
  FragBF a[16];
  {
    const unsigned short* arow = xn + (size_t)(rowBase + m) * D + hi * 8;
#pragma unroll
    for (int t = 0; t < 16; ++t) {
      a[t].q[0] = *(const u32x4*)(arow + t * 32);
      a[t].q[1] = *(const u32x4*)(arow + t * 32 + 16);
    }
  }

  float best[8];
  int   bidx[8];
#pragma unroll
  for (int r = 0; r < 8; ++r) { best[r] = -3.0e38f; bidx[r] = 0; }

#if VQ_HAVE_ASYNC_LDS
  // Issue one tile copy: 8 x async b128 per thread (32 KB per block),
  // tracked with ASYNCcnt; tile source is 16 consecutive codebook rows.
  auto issue_tile = [&](int kt, int buf) {
    const unsigned short* gsrc = cbn + (size_t)kt * 16 * D + tid * 8;
    unsigned short* lds = &tile[buf][tid * 8];
#pragma unroll
    for (int i = 0; i < 8; ++i) {
      __builtin_amdgcn_global_load_async_to_lds_b128(
          (VQ_AS_GLOBAL vq_i32x4*)(gsrc + i * 2048),
          (VQ_AS_LOCAL  vq_i32x4*)(lds  + i * 2048), 0, 0);
    }
  };
  int buf = 0;
  issue_tile(0, 0);
  vq_wait_async();
  __syncthreads();
#endif

  for (int kt = 0; kt < KCB / 16; ++kt) {
#if VQ_HAVE_ASYNC_LDS
    if (kt + 1 < KCB / 16) issue_tile(kt + 1, buf ^ 1);   // overlap next copy
#else
    const int buf = 0;
    __syncthreads();
    const u32x4* src = (const u32x4*)(cbn + (size_t)kt * 16 * D);
    u32x4* dst = (u32x4*)tile[0];
#pragma unroll
    for (int i = 0; i < 8; ++i) dst[tid + i * 256] = src[tid + i * 256];
    if (kt + 1 < KCB / 16)
      __builtin_prefetch(cbn + (size_t)(kt + 1) * 16 * D + tid * 32, 0, 0);
    __syncthreads();
#endif

    // B fragment: lane n holds code (n&15), d-chunks {d0+8*(n>>4)..+7,
    // d0+16+8*(n>>4)..+7} (same as A).  Accumulate all of D before the argmax compare.
    v8f acc = {};
    const unsigned short* brow = &tile[buf][m * D + hi * 8];
#pragma unroll
    for (int t = 0; t < 16; ++t) {
      FragBF b;
      b.q[0] = *(const u32x4*)(brow + t * 32);
      b.q[1] = *(const u32x4*)(brow + t * 32 + 16);
      acc = __builtin_amdgcn_wmma_f32_16x16x32_bf16(false, a[t].v, false, b.v,
                                                    (short)0, acc, false, false);
    }
    const int code = kt * 16 + m;
#pragma unroll
    for (int r = 0; r < 8; ++r) {           // strict > keeps lowest kt on ties
      bool gt = acc[r] > best[r];
      best[r] = gt ? acc[r] : best[r];
      bidx[r] = gt ? code : bidx[r];
    }

#if VQ_HAVE_ASYNC_LDS
    if (kt + 1 < KCB / 16) vq_wait_async();   // my async writes to buf^1 done
    __syncthreads();                          // everyone's done; reads of buf done
    buf ^= 1;
#endif
  }

  // Reduce the 16 candidate columns within each 16-lane half (xor of bits 0-3
  // never crosses the half boundary).  Prefer smaller index on exact ties.
#pragma unroll
  for (int off = 8; off; off >>= 1) {
#pragma unroll
    for (int r = 0; r < 8; ++r) {
      float ov = __shfl_xor(best[r], off, 32);
      int   oi = __shfl_xor(bidx[r], off, 32);
      bool take = (ov > best[r]) || (ov == best[r] && oi < bidx[r]);
      best[r] = take ? ov : best[r];
      bidx[r] = take ? oi : bidx[r];
    }
  }
  if (m == 0) {   // lanes 0 (rows 0..7) and 16 (rows 8..15)
#pragma unroll
    for (int r = 0; r < 8; ++r) {
      int row = rowBase + hi * 8 + r;
      idx_i[row] = bidx[r];
      idx_f[row] = (float)bidx[r];
    }
  }
}

// ---------------------------------------------------------------------------
// quantized = weight[idx]  (straight-through output == quantized) + MSE loss.
// One block per token.
// ---------------------------------------------------------------------------
__global__ __launch_bounds__(256) void vq_gather_loss(
    const float* __restrict__ inp, const float* __restrict__ weight,
    const int* __restrict__ idx, float* __restrict__ qout,
    float* __restrict__ loss) {
  __shared__ float red[8];
  const int n = blockIdx.x;
  const int k = idx[n];
  const float* w = weight + (size_t)k * D;
  const float* x = inp + (size_t)n * D;
  float* q = qout + (size_t)n * D;
  float part = 0.f;
#pragma unroll
  for (int i = 0; i < 2; ++i) {
    int d = threadIdx.x + i * 256;
    float wv = w[d];
    q[d] = wv;
    float df = wv - x[d];
    part += df * df;
  }
#pragma unroll
  for (int off = 16; off; off >>= 1) part += __shfl_xor(part, off, 32);
  if ((threadIdx.x & 31) == 0) red[threadIdx.x >> 5] = part;
  __syncthreads();
  if (threadIdx.x < 8) {
    float v = red[threadIdx.x];
#pragma unroll
    for (int off = 4; off; off >>= 1) v += __shfl_xor(v, off, 32);
    if (threadIdx.x == 0)
      unsafeAtomicAdd(loss, v * (1.0f / ((float)NTOK * (float)D)));
  }
}

// out_cs = ecs*DECAY, out_emaw = ema_w*DECAY, zero accumulators.
__global__ __launch_bounds__(256) void vq_init(
    const float* __restrict__ ecs, const float* __restrict__ emaw,
    float* __restrict__ out_cs, float* __restrict__ out_emaw,
    float* __restrict__ loss, float* __restrict__ nacc) {
  int i = blockIdx.x * 256 + threadIdx.x;
  if (i < KCB * D) out_emaw[i] = emaw[i] * DECAYF;
  if (i < KCB) out_cs[i] = ecs[i] * DECAYF;
  if (i == 0) { *loss = 0.f; *nacc = 0.f; }
}

// segment-sum via f32 atomics: out_emaw += 0.01*x, out_cs += 0.01 per token.
__global__ __launch_bounds__(256) void vq_scatter(
    const float* __restrict__ inp, const int* __restrict__ idx,
    float* __restrict__ out_cs, float* __restrict__ out_emaw) {
  int i = blockIdx.x * 256 + threadIdx.x;   // over NTOK*D
  int n = i >> 9;
  int d = i & (D - 1);
  int k = idx[n];
  unsafeAtomicAdd(&out_emaw[(size_t)k * D + d], OMDF * inp[i]);
  if (d == 0) unsafeAtomicAdd(&out_cs[k], OMDF);
}

__global__ __launch_bounds__(256) void vq_nsum(
    const float* __restrict__ out_cs, float* __restrict__ nacc) {
  __shared__ float red[8];
  float v = out_cs[blockIdx.x * 256 + threadIdx.x];
#pragma unroll
  for (int off = 16; off; off >>= 1) v += __shfl_xor(v, off, 32);
  if ((threadIdx.x & 31) == 0) red[threadIdx.x >> 5] = v;
  __syncthreads();
  if (threadIdx.x < 8) {
    float s = red[threadIdx.x];
#pragma unroll
    for (int off = 4; off; off >>= 1) s += __shfl_xor(s, off, 32);
    if (threadIdx.x == 0) unsafeAtomicAdd(nacc, s);
  }
}

// Laplace smoothing of cluster sizes + new_weight = new_ema_w / new_cs.
__global__ __launch_bounds__(256) void vq_finalize(
    float* __restrict__ out_cs, const float* __restrict__ nacc,
    const float* __restrict__ out_emaw, float* __restrict__ out_w) {
  const int k = blockIdx.x;
  float cs = out_cs[k];
  float nv = *nacc;
  float csf = (cs + EPSF) / (nv + (float)KCB * EPSF) * nv;
  __syncthreads();                 // all reads of out_cs[k] precede the write
  if (threadIdx.x == 0) out_cs[k] = csf;
  float inv = 1.0f / csf;
#pragma unroll
  for (int i = 0; i < 2; ++i) {
    int d = threadIdx.x + i * 256;
    out_w[(size_t)k * D + d] = out_emaw[(size_t)k * D + d] * inv;
  }
}

// ---------------------------------------------------------------------------
extern "C" void kernel_launch(void* const* d_in, const int* in_sizes, int n_in,
                              void* d_out, int out_size, void* d_ws, size_t ws_size,
                              hipStream_t stream) {
  const float* inp  = (const float*)d_in[0];   // [8,2048,512]
  const float* wgt  = (const float*)d_in[1];   // [K,D]
  const float* ecs  = (const float*)d_in[2];   // [K]
  const float* emaw = (const float*)d_in[3];   // [K,D]
  float* out = (float*)d_out;

  const size_t ND = (size_t)NTOK * D;          // 8,388,608
  const size_t KD = (size_t)KCB * D;           // 4,194,304
  float* out_q    = out;                       // quantized_st
  float* out_loss = out + ND;                  // scalar
  float* out_idx  = out_loss + 1;              // [NTOK] (as float)
  float* out_cs   = out_idx + NTOK;            // [K]
  float* out_emaw = out_cs + KCB;              // [K,D]
  float* out_w    = out_emaw + KD;             // [K,D]

  char* ws = (char*)d_ws;
  unsigned short* xn   = (unsigned short*)ws;                    // 16 MB bf16
  unsigned short* cbn  = (unsigned short*)(ws + ND * 2);         //  8 MB bf16
  int*            idx  = (int*)(ws + ND * 2 + KD * 2);           // 64 KB
  float*          nacc = (float*)(ws + ND * 2 + KD * 2 + (size_t)NTOK * 4);

  vq_init<<<(KCB * D + 255) / 256, 256, 0, stream>>>(ecs, emaw, out_cs, out_emaw,
                                                     out_loss, nacc);
  vq_normalize_bf16<<<NTOK / 8, 256, 0, stream>>>(inp, xn, NTOK);
  vq_normalize_bf16<<<KCB / 8, 256, 0, stream>>>(wgt, cbn, KCB);
  vq_argmax_wmma<<<NTOK / 128, 256, 0, stream>>>(xn, cbn, idx, out_idx);
  vq_gather_loss<<<NTOK, 256, 0, stream>>>(inp, wgt, idx, out_q, out_loss);
  vq_scatter<<<(int)(ND / 256), 256, 0, stream>>>(inp, idx, out_cs, out_emaw);
  vq_nsum<<<KCB / 256, 256, 0, stream>>>(out_cs, nacc);
  vq_finalize<<<KCB, 256, 0, stream>>>(out_cs, nacc, out_emaw, out_w);
}
